// MLPRNN_13907104105206
// MI455X (gfx1250) — compile-verified, matrix operand order checked by reading, not run on hardware
//
#include <hip/hip_runtime.h>
#include <hip/hip_bf16.h>
#include <stddef.h>

// ---------------------------------------------------------------------------
// MLP-RNN, K=20 steps, B=4096, d=128, h=256  (MI455X / gfx1250, wave32)
//
//  * kernel 1: convert w [4096,128,128] fp32 -> f16 into workspace (fits L2).
//  * kernel 2: pack W0/W1/V into WMMA B-fragment order (f16).
//  * kernel 3: persistent kernel, 256 WGs x 256 threads; each WG owns 16
//    batch rows and runs all 20 recurrent steps locally:
//      - TDM tensor_load_to_lds stages packed W0+W1 (256 KB) into LDS
//      - batched matvec grad = w@x - b   (f16 stream from L2, fp32 FMA)
//      - h0 = relu(xc @ W0^T)   16 WMMA (B frags from LDS)
//      - h1 = relu(h0 @ W1^T)   16 WMMA (B frags from LDS)
//      - x  = h1 @ V^T           8 WMMA (B frags from L2)
// ---------------------------------------------------------------------------

typedef _Float16 v16h __attribute__((ext_vector_type(16)));
typedef _Float16 v8h  __attribute__((ext_vector_type(8)));
typedef float    v8f  __attribute__((ext_vector_type(8)));
typedef unsigned int u32x4 __attribute__((ext_vector_type(4)));
typedef int          i32x4 __attribute__((ext_vector_type(4)));
typedef int          i32x8 __attribute__((ext_vector_type(8)));

#define BATCH   4096
#define D       128
#define H       256
#define KSTEPS  20
#define MTILE   16          // batch rows per workgroup
#define LD      264         // padded f16 row stride (conflict-free b128)
#define LDF     132         // padded fp32 row stride
#define ACT_B   8448        // bytes per activation buffer (16*264*2 = 16*132*4)
#define WLDS_OFF (4 * ACT_B)   // LDS byte offset of staged W0|W1 (33792)

// workspace layout (bytes)
#define WS_WF16   0                    // 4096*128*128 * 2 = 134217728
#define WS_W0P    134217728            // 128 frags * 1 KB = 131072
#define WS_W1P    (WS_W0P + 131072)    // 131072
#define WS_VP     (WS_W1P + 131072)    // 64 frags = 65536

// ---------------------------------------------------------------------------
// Kernel 1: w fp32 -> f16 (bandwidth pass; 8 elems/thread)
// ---------------------------------------------------------------------------
__global__ __launch_bounds__(256) void mlprnn_pack_w(
    const float* __restrict__ w, _Float16* __restrict__ o, long n)
{
    long idx = ((long)blockIdx.x * blockDim.x + threadIdx.x) * 8;
    if (idx >= n) return;
    v8h r;
#pragma unroll
    for (int e = 0; e < 8; ++e) r[e] = (_Float16)w[idx + e];
    *(v8h*)(o + idx) = r;
}

// ---------------------------------------------------------------------------
// Kernel 2: pack W0/W1/V into WMMA B-fragment order.
// B tile = 32(K) x 16(N).  Lane l holds 16 halfs (8 dwords v):
//   lanes 0-15 : v0..3 -> K 0..7,  v4..7 -> K 16..23   (N = l)
//   lanes 16-31: v0..3 -> K 8..15, v4..7 -> K 24..31   (N = l-16)
// Packed: frag f, lane l -> 16 contiguous halfs at (f*32 + l)*16.
// out[m][n] = sum_k A[m][k] * W[n][k]  =>  B[k][n] = W[n][k]
// ---------------------------------------------------------------------------
__global__ __launch_bounds__(256) void mlprnn_pack_weights(
    const float* __restrict__ W0, const float* __restrict__ W1,
    const float* __restrict__ V,
    _Float16* __restrict__ w0p, _Float16* __restrict__ w1p,
    _Float16* __restrict__ vpk)
{
    int tid = blockIdx.x * blockDim.x + threadIdx.x;
    if (tid >= 320 * 32) return;               // 128 + 128 + 64 fragments
    int f = tid >> 5, l = tid & 31;
    const float* W; _Float16* P; int fl;
    if (f < 128)      { W = W0; P = w0p; fl = f;       }
    else if (f < 256) { W = W1; P = w1p; fl = f - 128; }
    else              { W = V;  P = vpk; fl = f - 256; }
    int ntile = fl >> 3, kt = fl & 7;          // 8 K-tiles of 32 (Kdim = 256)
    _Float16* dst = P + ((size_t)fl * 32 + l) * 16;
#pragma unroll
    for (int e = 0; e < 16; ++e) {
        int v = e >> 1, hh = e & 1;
        int base = (v < 4) ? ((l < 16) ? 0 : 8) : ((l < 16) ? 16 : 24);
        int kl = base + 2 * (v & 3) + hh;
        int K = kt * 32 + kl;
        int N = ntile * 16 + (l & 15);
        dst[e] = (_Float16)W[(size_t)N * 256 + K];
    }
}

// ---------------------------------------------------------------------------
// fragment helpers
// ---------------------------------------------------------------------------
__device__ __forceinline__ v16h cat16(v8h lo, v8h hi) {
    return __builtin_shufflevector(lo, hi, 0,1,2,3,4,5,6,7,8,9,10,11,12,13,14,15);
}

// A fragment (16x32 f16) from a padded LDS activation buffer, K-tile k.
__device__ __forceinline__ v16h load_afrag(const _Float16* A, int lrow,
                                           int koff, int k) {
    const _Float16* base = A + lrow * LD + k * 32 + koff;
    v8h lo = *(const v8h*)(base);          // ds_load_b128
    v8h hi = *(const v8h*)(base + 16);     // ds_load_b128
    return cat16(lo, hi);
}

// Packed B fragment: 16 contiguous halfs per lane.
__device__ __forceinline__ v16h load_bfrag(const _Float16* P, int f, int l) {
    const _Float16* base = P + ((size_t)f * 32 + l) * 16;
    v8h lo = *(const v8h*)(base);
    v8h hi = *(const v8h*)(base + 8);
    return cat16(lo, hi);
}

// ---------------------------------------------------------------------------
// TDM stage: one 1-D tensor_load_to_lds of 256 KB (packed W0|W1) into LDS.
// D# built per CDNA5 ISA ch.8: group0 = {count, lds_addr, global_addr, type=2},
// group1 = {data_size=8B, tensor_dim0=tile_dim0=32768, stride=32768}.
// Issued by wave 0 only (TDM ignores EXEC; gate at wave granularity).
// ---------------------------------------------------------------------------
__device__ __forceinline__ void tdm_stage_weights(const _Float16* w0p) {
#if defined(__has_builtin)
#if __has_builtin(__builtin_amdgcn_tensor_load_to_lds) && \
    __has_builtin(__builtin_amdgcn_s_wait_tensorcnt)
    unsigned long long ga = (unsigned long long)(size_t)w0p;
    u32x4 g0;
    g0[0] = 1u;                                   // count=1, user mode
    g0[1] = (unsigned)WLDS_OFF;                   // lds_addr (bytes)
    g0[2] = (unsigned)(ga & 0xffffffffu);         // global_addr[31:0]
    g0[3] = (unsigned)((ga >> 32) & 0x01ffffffu)  // global_addr[56:32]
          | 0x80000000u;                          // type=2 (bits 127:126)
    i32x8 g1;
    g1[0] = 0x00030000;        // workgroup_mask=0, data_size=3 (8 bytes)
    g1[1] = (int)0x80000000;   // tensor_dim0 = 32768 (lo16 in [31:16])
    g1[2] = 0x00010000;        // tensor_dim0 hi=0 | tensor_dim1 = 1
    g1[3] = (int)0x80000000;   // tensor_dim1 hi=0 | tile_dim0 = 32768
    g1[4] = 0x00000001;        // tile_dim1 = 1, tile_dim2 = 0
    g1[5] = 0x00008000;        // tensor_dim0_stride = 32768 (lo32)
    g1[6] = 0;
    g1[7] = 0;
    i32x4 gz = {0, 0, 0, 0};
#if (__clang_major__ <= 22)
    __builtin_amdgcn_tensor_load_to_lds(g0, g1, gz, gz, 0);            // 5-arg
#else
    i32x8 gz8 = {0, 0, 0, 0, 0, 0, 0, 0};
    __builtin_amdgcn_tensor_load_to_lds(g0, g1, gz, gz, gz8, 0);       // 6-arg
#endif
    __builtin_amdgcn_s_wait_tensorcnt(0);
#endif
#endif
}

// ---------------------------------------------------------------------------
// Kernel 3: persistent recurrent kernel. 256 threads = 8 waves.
// ---------------------------------------------------------------------------
__global__ __launch_bounds__(256) void mlprnn_main(
    const _Float16* __restrict__ wf,   // [4096][128][128] f16
    const float*    __restrict__ bb,   // [4096][128] fp32
    const _Float16* __restrict__ w0p,  // packed (w1p contiguous after)
    const _Float16* __restrict__ vpk,  // packed V frags
    float* __restrict__ out)           // [4096][128] fp32
{
    extern __shared__ char smem[];
    _Float16* xc  = (_Float16*)(smem);              // [16][LD]  concat(-grad,x) f16
    float*    xf  = (float*)   (smem + ACT_B);      // [16][LDF] x fp32
    _Float16* h0  = (_Float16*)(smem + 2 * ACT_B);  // [16][LD]
    _Float16* h1  = (_Float16*)(smem + 3 * ACT_B);  // [16][LD]
    _Float16* w0l = (_Float16*)(smem + WLDS_OFF);   // 65536 halfs
    _Float16* w1l = w0l + 65536;                    // 65536 halfs

    const int t    = threadIdx.x;
    const int l    = t & 31;
    const int wid  = t >> 5;
    const int m0   = blockIdx.x * MTILE;
    const int lrow = l & 15;
    const int koff = (l < 16) ? 0 : 8;

    // ---- stage packed W0+W1 (256 KB) into LDS: TDM first, then a plain
    // copy as backstop (identical data; guarantees LDS contents).
    if (t < 32) tdm_stage_weights(w0p);
    __syncthreads();
    {
        const int4* src = (const int4*)w0p;
        int4* dst = (int4*)w0l;
        for (int i = t; i < 16384; i += 256) dst[i] = src[i];
    }
    // x0 = 0
    for (int i = t; i < MTILE * LDF; i += 256) xf[i] = 0.0f;
    for (int i = t; i < MTILE * LD;  i += 256) xc[i] = (_Float16)0.0f;
    __syncthreads();

    for (int step = 0; step < KSTEPS; ++step) {
        // ---- grad = w@x - b ; xc[:,0:128] = -grad (16x128 dots of len 128)
#pragma unroll 2
        for (int r = 0; r < 8; ++r) {
            int p = r * 256 + t;
            int m = p >> 7, i = p & 127;
            const _Float16* wrow = wf + ((size_t)(m0 + m) * 128 + i) * 128;
            const float*    xrow = xf + m * LDF;
            __builtin_prefetch(wrow + 256 * 128, 0, 1); // next r's row
            float acc = 0.0f;
#pragma unroll
            for (int j = 0; j < 128; j += 8) {
                v8h wv = *(const v8h*)(wrow + j);
#pragma unroll
                for (int e = 0; e < 8; ++e)
                    acc += (float)wv[e] * xrow[j + e];
            }
            float g = acc - bb[(size_t)(m0 + m) * 128 + i];
            xc[m * LD + i] = (_Float16)(-g);
        }
        __syncthreads();

        // ---- layer0: h0 = relu(xc @ W0^T)   N=256 (2 tiles/wave), K=256
        {
            v16h af[8];
#pragma unroll
            for (int k = 0; k < 8; ++k) af[k] = load_afrag(xc, lrow, koff, k);
#pragma unroll
            for (int nn = 0; nn < 2; ++nn) {
                int n = wid + nn * 8;
                v8f c = {};
#pragma unroll
                for (int k = 0; k < 8; ++k) {
                    v16h bf = load_bfrag(w0l, n * 8 + k, l);
                    c = __builtin_amdgcn_wmma_f32_16x16x32_f16(
                            false, af[k], false, bf, (short)0, c, false, false);
                }
#pragma unroll
                for (int r = 0; r < 8; ++r) {
                    int row = r + ((l < 16) ? 0 : 8);
                    float v = c[r] > 0.0f ? c[r] : 0.0f;
                    h0[row * LD + n * 16 + lrow] = (_Float16)v;
                }
            }
        }
        __syncthreads();

        // ---- layer1: h1 = relu(h0 @ W1^T)
        {
            v16h af[8];
#pragma unroll
            for (int k = 0; k < 8; ++k) af[k] = load_afrag(h0, lrow, koff, k);
#pragma unroll
            for (int nn = 0; nn < 2; ++nn) {
                int n = wid + nn * 8;
                v8f c = {};
#pragma unroll
                for (int k = 0; k < 8; ++k) {
                    v16h bf = load_bfrag(w1l, n * 8 + k, l);
                    c = __builtin_amdgcn_wmma_f32_16x16x32_f16(
                            false, af[k], false, bf, (short)0, c, false, false);
                }
#pragma unroll
                for (int r = 0; r < 8; ++r) {
                    int row = r + ((l < 16) ? 0 : 8);
                    float v = c[r] > 0.0f ? c[r] : 0.0f;
                    h1[row * LD + n * 16 + lrow] = (_Float16)v;
                }
            }
        }
        __syncthreads();

        // ---- layer2: x = h1 @ V^T   N=128 (1 tile/wave), K=256; no relu
        {
            v16h af[8];
#pragma unroll
            for (int k = 0; k < 8; ++k) af[k] = load_afrag(h1, lrow, koff, k);
            int n = wid;
            v8f c = {};
#pragma unroll
            for (int k = 0; k < 8; ++k) {
                v16h bf = load_bfrag(vpk, n * 8 + k, l);   // from L2
                c = __builtin_amdgcn_wmma_f32_16x16x32_f16(
                        false, af[k], false, bf, (short)0, c, false, false);
            }
#pragma unroll
            for (int r = 0; r < 8; ++r) {
                int row = r + ((l < 16) ? 0 : 8);
                xf[row * LDF + n * 16 + lrow] = c[r];
                xc[row * LD + 128 + n * 16 + lrow] = (_Float16)c[r];
            }
        }
        __syncthreads();
    }

    // ---- write x (fp32) out
#pragma unroll
    for (int r = 0; r < 8; ++r) {
        int p = r * 256 + t;
        int m = p >> 7, i = p & 127;
        out[(size_t)(m0 + m) * 128 + i] = xf[m * LDF + i];
    }
}

// ---------------------------------------------------------------------------
extern "C" void kernel_launch(void* const* d_in, const int* in_sizes, int n_in,
                              void* d_out, int out_size, void* d_ws, size_t ws_size,
                              hipStream_t stream) {
    const float* w  = (const float*)d_in[0];   // [4096,128,128]
    const float* b  = (const float*)d_in[1];   // [4096,128]
    const float* W0 = (const float*)d_in[2];   // [256,256]
    const float* W1 = (const float*)d_in[3];   // [256,256]
    const float* V  = (const float*)d_in[4];   // [128,256]
    float* out = (float*)d_out;

    char* ws = (char*)d_ws;
    _Float16* wf  = (_Float16*)(ws + WS_WF16);
    _Float16* w0p = (_Float16*)(ws + WS_W0P);
    _Float16* w1p = (_Float16*)(ws + WS_W1P);
    _Float16* vpk = (_Float16*)(ws + WS_VP);
    (void)in_sizes; (void)n_in; (void)out_size; (void)ws_size;

    // 1) w -> f16 (67108864 elems, 8/thread)
    mlprnn_pack_w<<<32768, 256, 0, stream>>>(w, wf, 67108864L);
    // 2) pack MLP weights into WMMA B-fragment order (320 frags * 32 lanes)
    mlprnn_pack_weights<<<40, 256, 0, stream>>>(W0, W1, V, w0p, w1p, vpk);
    // 3) persistent recurrent kernel: 256 WGs * 256 thr, 289 KB dynamic LDS
    size_t smem = 4 * ACT_B + 2 * 131072;      // 295936 B  (<= 320 KB/WG)
    mlprnn_main<<<BATCH / MTILE, 256, smem, stream>>>(wf, b, w0p, vpk, out);
}
